// ContrastiveLossModule_82386062672242
// MI455X (gfx1250) — matrix operand order repaired and checked by reading, not compiled, not run on hardware
//
#include <hip/hip_runtime.h>
#include <hip/hip_bf16.h>
#include <stdint.h>

// ---------------------------------------------------------------------------
// CDNA5 (gfx1250) wave32 WMMA implementation of the conv-encoder +
// contrastive loss reference.  All matrix math on v_wmma_f32_16x16x32_f16.
// Convs are tap-decomposed implicit GEMMs with channel-fastest LDS tiles so
// B-fragments are contiguous ds_load_b128 pairs.  conv2/conv3 input staging
// uses the Tensor Data Mover (tensor_load_to_lds + s_wait_tensorcnt).
// ---------------------------------------------------------------------------

typedef __attribute__((ext_vector_type(16))) _Float16 v16h;
typedef __attribute__((ext_vector_type(8)))  _Float16 v8h;
typedef __attribute__((ext_vector_type(8)))  float    v8f;

#if defined(__gfx1250__) && __has_builtin(__builtin_amdgcn_tensor_load_to_lds) && \
    __has_builtin(__builtin_amdgcn_s_wait_tensorcnt)
#define HAVE_TDM 1
#else
#define HAVE_TDM 0
#endif

#if HAVE_TDM
typedef __attribute__((ext_vector_type(4))) unsigned int tdm_v4u;
typedef __attribute__((ext_vector_type(8))) int          tdm_v8i;
typedef __attribute__((ext_vector_type(4))) int          tdm_v4i;

// 2-D TDM load: nrows x row_elems f16 tile, global rows contiguous
// (stride row_stride_elems), into LDS with pad_amount dwords appended after
// every 2^(pad_interval_code+1) dwords (i.e. after each row when the row is a
// power-of-two number of dwords).  D# packing per CDNA5 ISA sec 8.3/8.4.
__device__ __forceinline__ void tdm_load_2d_to_lds(const void* gaddr,
                                                   unsigned lds_byte_off,
                                                   int row_elems, int nrows,
                                                   int row_stride_elems,
                                                   int pad_interval_code,
                                                   int pad_amount_code) {
  unsigned long long ga = (unsigned long long)gaddr;
  tdm_v4u g0;
  g0[0] = 1u;                                   // count=1, user descriptor
  g0[1] = lds_byte_off;                         // lds_addr (bytes)
  g0[2] = (unsigned)(ga & 0xffffffffu);         // global_addr[31:0]
  g0[3] = (unsigned)((ga >> 32) & 0x1ffffffu)   // global_addr[56:32]
        | (2u << 30);                           // type = 2 ("image")
  tdm_v8i g1;
  g1[0] = (int)((1u << 16)                      // data_size = 1 (2 bytes)
              | (1u << 20)                      // pad_enable
              | ((unsigned)pad_interval_code << 22)
              | ((unsigned)pad_amount_code << 25));
  g1[1] = (int)(((unsigned)row_elems & 0xffffu) << 16);   // td0[15:0] (atomic_addr=0)
  g1[2] = (int)((((unsigned)row_elems >> 16) & 0xffffu)   // td0[31:16]
              | (((unsigned)nrows & 0xffffu) << 16));     // td1[15:0]
  g1[3] = (int)((((unsigned)nrows >> 16) & 0xffffu)       // td1[31:16]
              | (((unsigned)row_elems & 0xffffu) << 16)); // tile_dim0
  g1[4] = (int)((unsigned)nrows & 0xffffu);               // tile_dim1 (tile_dim2=0)
  g1[5] = (int)(unsigned)row_stride_elems;                // td0_stride[31:0]
  g1[6] = 0;                                              // td0_stride hi / td1_stride lo
  g1[7] = 0;                                              // td1_stride hi
  tdm_v4i z4 = {0, 0, 0, 0};
#if __clang_major__ >= 23
  tdm_v8i z8 = {0, 0, 0, 0, 0, 0, 0, 0};
  __builtin_amdgcn_tensor_load_to_lds(g0, g1, z4, z4, z8, 0);
#else
  __builtin_amdgcn_tensor_load_to_lds(g0, g1, z4, z4, 0);
#endif
}
#endif  // HAVE_TDM

__device__ __forceinline__ v8f wmma_f16f32(v16h a, v16h b, v8f c) {
  return __builtin_amdgcn_wmma_f32_16x16x32_f16(false, a, false, b, (short)0, c,
                                                false, false);
}

// A-fragment (16x32, MxK) from f16 row-major [M][Kstride].
__device__ __forceinline__ v16h load_frag_a(const _Float16* __restrict__ W, int mrow,
                                            int Kstride, int kbase, int lane) {
  const int k0 = kbase + ((lane & 16) ? 16 : 0);
  const _Float16* p = W + (size_t)mrow * (size_t)Kstride + k0;
  union { v16h v; v8h h[2]; } u;
  u.h[0] = *(const v8h*)(p);
  u.h[1] = *(const v8h*)(p + 8);
  return u.v;
}

// B-fragment (32x16, KxN), column n = row `nrow` of row-major [N][Kstride] f16.
__device__ __forceinline__ v16h load_frag_bT(const _Float16* __restrict__ X, int nrow,
                                             int Kstride, int kbase, int lane) {
  const int k0 = kbase + ((lane & 16) ? 16 : 0);
  return *(const v16h*)(X + (size_t)nrow * (size_t)Kstride + k0);
}

// ---------------------------------------------------------------------------
// Conversion / repack kernels (run once per launch; negligible cost)
// ---------------------------------------------------------------------------
__global__ void cvt_scale_f16_kernel(const float* __restrict__ in,
                                     _Float16* __restrict__ out,
                                     float scale, int n) {
  int i = blockIdx.x * blockDim.x + threadIdx.x;
  if (i < n) out[i] = (_Float16)(in[i] * scale);
}

__global__ void cvt_f16_rows_kernel(const float* __restrict__ in,
                                    _Float16* __restrict__ out, int total) {
  int i = blockIdx.x * blockDim.x + threadIdx.x;
  if (i < total) out[i] = (_Float16)in[i];
}

// conv weights OIHW f32 [O][CIN][3][3] -> f16 [O][9][CINPAD]
__global__ void conv_w_repack_kernel(const float* __restrict__ w,
                                     _Float16* __restrict__ out,
                                     int CIN, int CINPAD, int total) {
  int i = blockIdx.x * blockDim.x + threadIdx.x;
  if (i >= total) return;
  int o = i / (9 * CINPAD);
  int rem = i - o * 9 * CINPAD;
  int rs = rem / CINPAD;
  int c = rem - rs * CINPAD;
  out[i] = (c < CIN) ? (_Float16)w[((size_t)o * CIN + c) * 9 + rs] : (_Float16)0.f;
}

// fc1 weights [512][9216] k=c*36+n  ->  f16 [512][9216] k'=n*256+c
__global__ void fc1_w_repack_kernel(const float* __restrict__ w,
                                    _Float16* __restrict__ out) {
  int i = blockIdx.x * blockDim.x + threadIdx.x;
  if (i >= 512 * 9216) return;
  int j = i / 9216;
  int kp = i - j * 9216;
  int n = kp >> 8, c = kp & 255;
  out[i] = (_Float16)w[(size_t)j * 9216 + c * 36 + n];
}

// ---------------------------------------------------------------------------
// Tap-decomposed implicit-GEMM conv (pad=1), fused bias+ReLU.
// Input  x: TSTAGE ? [B][CIN][H*W] : [B][H*W][CIN]   (f16)
// Weights wrs: [COUT][9][CINPAD];  Output y: [B][OH*OW][COUT]
// LDS tile [PH*PW][CROW], channel-fastest, zero borders & channel pad.
// ---------------------------------------------------------------------------
template <int CIN, int H, int W, int COUT, int OH, int OW, int STRIDE,
          int CINPAD, int CROW, int NTILES, int MGROUPS, int MPW, bool TSTAGE>
__global__ void conv_wmma_kernel(const _Float16* __restrict__ x,
                                 const _Float16* __restrict__ wrs,
                                 const float* __restrict__ bias,
                                 _Float16* __restrict__ y) {
  constexpr int PH = H + 2, PW = W + 2;
  constexpr int N = OH * OW;
  constexpr int KS = 9 * CINPAD;

  const int b = blockIdx.x;
  const int tid = threadIdx.x;
  const int wid = tid >> 5;
  const int lane = tid & 31;

  __shared__ _Float16 tile[PH * PW * CROW];

  for (int i = tid; i < PH * PW * CROW; i += 256) tile[i] = (_Float16)0.f;
  __syncthreads();

  bool staged = false;
#if HAVE_TDM
  // TDM path: rows are CIN f16 (power-of-two dwords), channel pad appended
  // after each row by the TDM pad engine.  One descriptor per spatial row.
  if constexpr (!TSTAGE && ((CIN & (CIN - 1)) == 0) && (CROW > CIN)) {
    constexpr int kPadIvl = __builtin_ctz(CIN) - 2;        // 2^(c+1) dwords = CIN/2
    constexpr int kPadAmt = (CROW - CIN) / 2 - 1;          // dwords-1 encoding
    if (wid == 0) {
      const _Float16* src = x + (size_t)b * (H * W * CIN);
      const unsigned lds0 = (unsigned)(uintptr_t)(&tile[0]);
      for (int yy = 0; yy < H; ++yy) {
        tdm_load_2d_to_lds(src + (size_t)yy * W * CIN,
                           lds0 + (unsigned)(((yy + 1) * PW + 1) * CROW) * 2u,
                           CIN, W, CIN, kPadIvl, kPadAmt);
      }
      __builtin_amdgcn_s_wait_tensorcnt(0);
    }
    staged = true;
  }
#endif
  if (!staged) {
    if constexpr (TSTAGE) {
      const _Float16* src = x + (size_t)b * (CIN * H * W);  // [c][sp]
      for (int i = tid; i < CIN * H * W; i += 256) {
        int c = i / (H * W);
        int sp = i - c * (H * W);
        int yy = sp / W, xx = sp - yy * W;
        tile[((yy + 1) * PW + xx + 1) * CROW + c] = src[i];
      }
    } else {
      const _Float16* src = x + (size_t)b * (H * W * CIN);  // [sp][c]
      for (int i = tid; i < H * W * CIN; i += 256) {
        int sp = i / CIN;
        int c = i - sp * CIN;
        int yy = sp / W, xx = sp - yy * W;
        tile[((yy + 1) * PW + xx + 1) * CROW + c] = src[i];
      }
    }
  }
  __syncthreads();

  if (wid >= NTILES * MGROUPS) return;
  const int ntile = wid % NTILES;
  const int mtb = (wid / NTILES) * MPW;
  const int nlo = lane & 15;
  const int khalf = (lane & 16) ? 16 : 0;
  const int ncol = ntile * 16 + nlo;
  const bool nvalid = ncol < N;
  int oy = ncol / OW, ox = ncol - (ncol / OW) * OW;
  if (!nvalid) { oy = 0; ox = 0; }

  const v8f zero = {};
  v8f acc[MPW];
#pragma unroll
  for (int m = 0; m < MPW; ++m) acc[m] = zero;

  for (int rs = 0; rs < 9; ++rs) {
    const int r = rs / 3, s = rs - (rs / 3) * 3;
    const _Float16* bp =
        &tile[((oy * STRIDE + r) * PW + (ox * STRIDE + s)) * CROW + khalf];
#pragma unroll
    for (int kb = 0; kb < CINPAD; kb += 32) {
      union { v16h v; v8h h[2]; } bu;   // contiguous 16 halves: ds_load_b128 x2
      bu.h[0] = *(const v8h*)(bp + kb);
      bu.h[1] = *(const v8h*)(bp + kb + 8);
      const int kbase = rs * CINPAD + kb;
#pragma unroll
      for (int m = 0; m < MPW; ++m) {
        v16h a = load_frag_a(wrs, (mtb + m) * 16 + nlo, KS, kbase, lane);
        acc[m] = wmma_f16f32(a, bu.v, acc[m]);
      }
    }
  }

  if (nvalid) {
    _Float16* dst = y + ((size_t)b * N + ncol) * COUT;
    const int mofs = (lane & 16) ? 8 : 0;
#pragma unroll
    for (int m = 0; m < MPW; ++m) {
      const int mb = (mtb + m) * 16 + mofs;
      union { v8h v; _Float16 e[8]; } o;
#pragma unroll
      for (int v = 0; v < 8; ++v) {
        float t = acc[m][v] + bias[mb + v];
        o.e[v] = (_Float16)(t > 0.f ? t : 0.f);
      }
      *(v8h*)(dst + mb) = o.v;  // contiguous 16B store
    }
  }
}

// ---------------------------------------------------------------------------
// GEMM: Out[n][m] = sum_k X[n][k] * Wf[m][k] + bias[m]   (f32 output [N][M])
// One wave: one mtile x 4 ntiles (A fragment reused 4x).  No prefetch in the
// K-loop: it breaks the compiler's 4x unroll + pipelined partial waits.
// ---------------------------------------------------------------------------
__global__ void gemm_nt_wmma_kernel(const _Float16* __restrict__ Wf,  // [M][K]
                                    const _Float16* __restrict__ X,   // [N][K]
                                    const float* __restrict__ bias,   // [M]
                                    float* __restrict__ Out,          // [N][M]
                                    int M, int Nn, int K) {
  const int tid = threadIdx.x;
  const int wid = tid >> 5;
  const int lane = tid & 31;
  const int gw = blockIdx.x * 8 + wid;
  const int MT = M >> 4;
  const int mtile = gw % MT;
  const int ngrp = gw / MT;
  const int nbase = ngrp * 64;
  if (nbase >= Nn) return;

  const int nlo = lane & 15;
  const int mrow = mtile * 16 + nlo;
  v8f acc[4] = {{}, {}, {}, {}};
  for (int kb = 0; kb < K; kb += 32) {
    v16h a = load_frag_a(Wf, mrow, K, kb, lane);
#pragma unroll
    for (int j = 0; j < 4; ++j) {
      v16h bf = load_frag_bT(X, nbase + j * 16 + nlo, K, kb, lane);
      acc[j] = wmma_f16f32(a, bf, acc[j]);
    }
  }
  const int mbase = mtile * 16 + ((lane & 16) ? 8 : 0);
#pragma unroll
  for (int j = 0; j < 4; ++j) {
    int ncol = nbase + j * 16 + nlo;
#pragma unroll
    for (int v = 0; v < 8; ++v) {
      int m = mbase + v;
      Out[(size_t)ncol * M + m] = acc[j][v] + bias[m];
    }
  }
}

// ---------------------------------------------------------------------------
// LayerNorm + SiLU, f32 in -> f16 out.  One wave per row.
// ---------------------------------------------------------------------------
__global__ void ln_silu_kernel(const float* __restrict__ in,
                               const float* __restrict__ g,
                               const float* __restrict__ bb,
                               _Float16* __restrict__ out, int D) {
  const int wid = threadIdx.x >> 5, lane = threadIdx.x & 31;
  const int row = blockIdx.x * 8 + wid;
  const float* x = in + (size_t)row * D;
  float s = 0.f, s2 = 0.f;
  for (int i = lane; i < D; i += 32) {
    float v = x[i];
    s += v;
    s2 += v * v;
  }
#pragma unroll
  for (int m = 16; m; m >>= 1) {
    s += __shfl_xor(s, m, 32);
    s2 += __shfl_xor(s2, m, 32);
  }
  float mean = s / (float)D;
  float var = s2 / (float)D - mean * mean;
  float rstd = rsqrtf(var + 1e-5f);
  _Float16* o = out + (size_t)row * D;
  for (int i = lane; i < D; i += 32) {
    float v = (x[i] - mean) * rstd * g[i] + bb[i];
    float sig = 1.f / (1.f + __expf(-v));
    o[i] = (_Float16)(v * sig);
  }
}

// ---------------------------------------------------------------------------
// emb f32 [4096][64] -> f16 copy + squared row norms.  One wave per row.
// ---------------------------------------------------------------------------
__global__ void emb_prep_kernel(const float* __restrict__ emb,
                                _Float16* __restrict__ embh,
                                float* __restrict__ norms) {
  const int wid = threadIdx.x >> 5, lane = threadIdx.x & 31;
  const int row = blockIdx.x * 8 + wid;
  const float* e = emb + (size_t)row * 64;
  float s = 0.f;
  for (int i = lane; i < 64; i += 32) {
    float v = e[i];
    s += v * v;
    embh[(size_t)row * 64 + i] = (_Float16)v;
  }
#pragma unroll
  for (int m = 16; m; m >>= 1) s += __shfl_xor(s, m, 32);
  if (lane == 0) norms[row] = s;
}

// ---------------------------------------------------------------------------
// Contrastive loss.  Deterministic per-block partials.
// ---------------------------------------------------------------------------
__global__ void loss_kernel(const _Float16* __restrict__ embh,  // [4096][64]
                            const float* __restrict__ norms,    // [4096]
                            float* __restrict__ part) {         // [gridDim][2]
  const int tid = threadIdx.x, wid = tid >> 5, lane = tid & 31;
  const int atile = blockIdx.x;
  __shared__ float smax[8][16];
  __shared__ float ssum[8][16];
  __shared__ float sdiag[16];

  const int nlo = lane & 15;
  const int mrel_lo = (lane & 16) ? 8 : 0;
  const int ibase = atile * 16;

  float mx[8], sm[8], na[8];
#pragma unroll
  for (int v = 0; v < 8; ++v) {
    mx[v] = -1e30f;
    sm[v] = 0.f;
    na[v] = norms[ibase + mrel_lo + v];
  }

  const _Float16* A = embh;
  const _Float16* T = embh + (size_t)2048 * 64;

  const int mrow = ibase + nlo;
  for (int nt = wid; nt < 128; nt += 8) {
    const int nrow = nt * 16 + nlo;
    v8f dot = {};
#pragma unroll
    for (int kb = 0; kb < 64; kb += 32) {
      v16h a = load_frag_a(A, mrow, 64, kb, lane);
      v16h b = load_frag_bT(T, nrow, 64, kb, lane);
      dot = wmma_f16f32(a, b, dot);
    }
    const int j = nt * 16 + nlo;
    const float ntj = norms[2048 + j];
#pragma unroll
    for (int v = 0; v < 8; ++v) {
      int i = ibase + mrel_lo + v;
      float d2 = na[v] + ntj - 2.f * dot[v];
      float lg = -sqrtf(fmaxf(d2, 0.f) + 1e-8f);
      if (i == j) sdiag[mrel_lo + v] = lg;
      float m2 = fmaxf(mx[v], lg);
      sm[v] = sm[v] * __expf(mx[v] - m2) + __expf(lg - m2);
      mx[v] = m2;
    }
  }
#pragma unroll
  for (int msk = 1; msk < 16; msk <<= 1) {
#pragma unroll
    for (int v = 0; v < 8; ++v) {
      float om = __shfl_xor(mx[v], msk, 32);
      float os = __shfl_xor(sm[v], msk, 32);
      float m2 = fmaxf(mx[v], om);
      sm[v] = sm[v] * __expf(mx[v] - m2) + os * __expf(om - m2);
      mx[v] = m2;
    }
  }
  if (nlo == 0) {
#pragma unroll
    for (int v = 0; v < 8; ++v) {
      smax[wid][mrel_lo + v] = mx[v];
      ssum[wid][mrel_lo + v] = sm[v];
    }
  }
  __syncthreads();

  if (tid < 16) {
    float m = -1e30f, s = 0.f;
#pragma unroll
    for (int w2 = 0; w2 < 8; ++w2) {
      float om = smax[w2][tid], os = ssum[w2][tid];
      float m2 = fmaxf(m, om);
      s = s * __expf(m - m2) + os * __expf(om - m2);
      m = m2;
    }
    float lse = m + __logf(s);
    float contrib = sdiag[tid] - lse;
    float l2 = lse * lse;
#pragma unroll
    for (int msk = 1; msk < 16; msk <<= 1) {
      contrib += __shfl_xor(contrib, msk, 32);
      l2 += __shfl_xor(l2, msk, 32);
    }
    if (tid == 0) {
      part[blockIdx.x * 2 + 0] = contrib;
      part[blockIdx.x * 2 + 1] = l2;
    }
  }
}

__global__ void finalize_kernel(const float* __restrict__ part,
                                float* __restrict__ out) {
  float sumA = 0.f, sumL2 = 0.f;
  for (int i = 0; i < 128; ++i) {
    sumA += part[i * 2 + 0];
    sumL2 += part[i * 2 + 1];
  }
  float infonce = -(sumA / 2048.f);
  float reg = 0.1f * (sumL2 / 2048.f);
  out[0] = (infonce + reg) * 0.1f;
}

// ---------------------------------------------------------------------------
// Host launcher
// ---------------------------------------------------------------------------
extern "C" void kernel_launch(void* const* d_in, const int* in_sizes, int n_in,
                              void* d_out, int out_size, void* d_ws, size_t ws_size,
                              hipStream_t stream) {
  (void)in_sizes; (void)n_in; (void)out_size; (void)ws_size;

  const float* obs   = (const float*)d_in[0];
  const float* w1    = (const float*)d_in[2];
  const float* b1    = (const float*)d_in[3];
  const float* w2    = (const float*)d_in[4];
  const float* b2    = (const float*)d_in[5];
  const float* w3    = (const float*)d_in[6];
  const float* b3    = (const float*)d_in[7];
  const float* fc1_w = (const float*)d_in[8];
  const float* fc1_b = (const float*)d_in[9];
  const float* ln1_g = (const float*)d_in[10];
  const float* ln1_b = (const float*)d_in[11];
  const float* fc2_w = (const float*)d_in[12];
  const float* fc2_b = (const float*)d_in[13];
  const float* ln2_g = (const float*)d_in[14];
  const float* ln2_b = (const float*)d_in[15];
  const float* out_w = (const float*)d_in[16];
  const float* out_b = (const float*)d_in[17];

  enum { NB = 4096, C0 = 84, SP = 121, C1 = 64, C2 = 128, C3 = 256,
         FLAT = 9216, HID = 512, EMBD = 64 };

  char* base = (char*)d_ws;
  size_t off = 0;
  auto balloc = [&](size_t bytes) -> char* {
    char* p = base + off;
    off = (off + bytes + 255) & ~(size_t)255;
    return p;
  };

  const size_t x0_elems = (size_t)NB * C0 * SP;
  const size_t x3_elems = (size_t)NB * FLAT;
  _Float16* x0    = (_Float16*)balloc(sizeof(_Float16) *
                      (x0_elems > x3_elems ? x0_elems : x3_elems));
  _Float16* x3    = x0;  // alias: x0 dead after conv1
  _Float16* w1p   = (_Float16*)balloc(sizeof(_Float16) * (size_t)C1 * 9 * 96);
  _Float16* w2p   = (_Float16*)balloc(sizeof(_Float16) * (size_t)C2 * 9 * 64);
  _Float16* w3p   = (_Float16*)balloc(sizeof(_Float16) * (size_t)C3 * 9 * 128);
  _Float16* fc1p  = (_Float16*)balloc(sizeof(_Float16) * (size_t)HID * FLAT);
  _Float16* fc2p  = (_Float16*)balloc(sizeof(_Float16) * (size_t)HID * HID);
  _Float16* outp  = (_Float16*)balloc(sizeof(_Float16) * (size_t)EMBD * HID);
  _Float16* x1    = (_Float16*)balloc(sizeof(_Float16) * (size_t)NB * SP * C1);
  _Float16* x2    = (_Float16*)balloc(sizeof(_Float16) * (size_t)NB * SP * C2);
  float*    h1raw = (float*)balloc(sizeof(float) * (size_t)NB * HID);
  float*    h2raw = h1raw;
  _Float16* h1h   = (_Float16*)balloc(sizeof(_Float16) * (size_t)NB * HID);
  _Float16* h2h   = h1h;
  float*    emb   = (float*)balloc(sizeof(float) * (size_t)NB * EMBD);
  _Float16* embh  = (_Float16*)balloc(sizeof(_Float16) * (size_t)NB * EMBD);
  float*    norms = (float*)balloc(sizeof(float) * NB);
  float*    part  = (float*)balloc(sizeof(float) * 128 * 2);

  {
    int n = (int)x0_elems;
    cvt_scale_f16_kernel<<<(n + 255) / 256, 256, 0, stream>>>(obs, x0, 1.f / 255.f, n);
  }
  {
    int t;
    t = C1 * 9 * 96;   conv_w_repack_kernel<<<(t + 255) / 256, 256, 0, stream>>>(w1, w1p, 84, 96, t);
    t = C2 * 9 * 64;   conv_w_repack_kernel<<<(t + 255) / 256, 256, 0, stream>>>(w2, w2p, 64, 64, t);
    t = C3 * 9 * 128;  conv_w_repack_kernel<<<(t + 255) / 256, 256, 0, stream>>>(w3, w3p, 128, 128, t);
    t = HID * FLAT;    fc1_w_repack_kernel<<<(t + 255) / 256, 256, 0, stream>>>(fc1_w, fc1p);
    t = HID * HID;     cvt_f16_rows_kernel<<<(t + 255) / 256, 256, 0, stream>>>(fc2_w, fc2p, t);
    t = EMBD * HID;    cvt_f16_rows_kernel<<<(t + 255) / 256, 256, 0, stream>>>(out_w, outp, t);
  }
  conv_wmma_kernel<84, 11, 11, 64, 11, 11, 1, 96, 104, 8, 1, 4, true>
      <<<NB, 256, 0, stream>>>(x0, w1p, b1, x1);
  conv_wmma_kernel<64, 11, 11, 128, 11, 11, 1, 64, 72, 8, 1, 8, false>
      <<<NB, 256, 0, stream>>>(x1, w2p, b2, x2);
  conv_wmma_kernel<128, 11, 11, 256, 6, 6, 2, 128, 136, 3, 2, 8, false>
      <<<NB, 256, 0, stream>>>(x2, w3p, b3, x3);
  gemm_nt_wmma_kernel<<<(HID / 16) * (NB / 64) / 8, 256, 0, stream>>>(fc1p, x3, fc1_b, h1raw, HID, NB, FLAT);
  ln_silu_kernel<<<NB / 8, 256, 0, stream>>>(h1raw, ln1_g, ln1_b, h1h, HID);
  gemm_nt_wmma_kernel<<<(HID / 16) * (NB / 64) / 8, 256, 0, stream>>>(fc2p, h1h, fc2_b, h2raw, HID, NB, HID);
  ln_silu_kernel<<<NB / 8, 256, 0, stream>>>(h2raw, ln2_g, ln2_b, h2h, HID);
  gemm_nt_wmma_kernel<<<(EMBD / 16) * (NB / 64) / 8, 256, 0, stream>>>(outp, h2h, out_b, emb, EMBD, NB, HID);
  emb_prep_kernel<<<NB / 8, 256, 0, stream>>>(emb, embh, norms);
  loss_kernel<<<128, 256, 0, stream>>>(embh, norms, part);
  finalize_kernel<<<1, 1, 0, stream>>>(part, (float*)d_out);
}
